// TContrastive_66864050864739
// MI455X (gfx1250) — matrix operand order repaired and checked by reading, not compiled, not run on hardware
//
#include <hip/hip_runtime.h>
#include <hip/hip_bf16.h>

typedef __attribute__((ext_vector_type(16))) __bf16 v16bf;
typedef __attribute__((ext_vector_type(8)))  float  v8f;

#define NROWS 8192
#define DIM   1024
#define LDS_S 40              // padded LDS row stride (bf16 elems), 16B-aligned
#define TAU_INV 2.0f          // 1 / 0.5

union Frag { v16bf v; uint4 q[2]; };
union Pk4  { __bf16 h[4]; uint2 u; };

// ---------------------------------------------------------------- async copy
// gfx1250 async DMA: global -> LDS, tracked by ASYNCcnt (no VGPR data path).
__device__ __forceinline__ void async_b128(const __bf16* g, __bf16* l) {
  asm volatile("global_load_async_to_lds_b128 %0, %1, off"
               :: "v"((__attribute__((address_space(3))) char*)l), "v"(g)
               : "memory");
}
__device__ __forceinline__ void wait_async0() {
  asm volatile("s_wait_asynccnt 0x0" ::: "memory");
}

// ---------------------------------------------------------------- converts
__global__ void convert_f32_bf16(const float* __restrict__ src,
                                 __bf16* __restrict__ dst, int n4) {
  int i = blockIdx.x * blockDim.x + threadIdx.x;
  if (i < n4) {
    float4 f = reinterpret_cast<const float4*>(src)[i];
    Pk4 p;
    p.h[0] = (__bf16)f.x; p.h[1] = (__bf16)f.y;
    p.h[2] = (__bf16)f.z; p.h[3] = (__bf16)f.w;
    reinterpret_cast<uint2*>(dst)[i] = p.u;
  }
}

// dst[n][b*32+a] = src[n][a*32+b]   (per-sample 32x32 transpose, column view)
__global__ __launch_bounds__(256) void convert_transpose32(
    const float* __restrict__ src, __bf16* __restrict__ dst) {
  __shared__ float t[32][33];
  size_t base = (size_t)blockIdx.x * 1024;
  for (int i = threadIdx.x; i < 1024; i += 256)
    t[i >> 5][i & 31] = src[base + i];
  __syncthreads();
  for (int i = threadIdx.x; i < 1024; i += 256) {
    int b = i >> 5, a = i & 31;
    dst[base + i] = (__bf16)t[a][b];
  }
}

// ---------------------------------------------------------------- fragments
// A fragment: 16x32 bf16, lane&15 = M row; K chunks {kb..kb+8} and {kb+16..kb+24},
// kb = (lane>>4)*8  (per CDNA5 ISA 16-bit A layout)
__device__ __forceinline__ v16bf load_a_frag(const __bf16* lA, int lane, int wrow) {
  Frag f;
  int row = wrow + (lane & 15);
  int kb  = (lane >> 4) * 8;
  f.q[0] = *reinterpret_cast<const uint4*>(lA + row * LDS_S + kb);
  f.q[1] = *reinterpret_cast<const uint4*>(lA + row * LDS_S + kb + 16);
  return f.v;
}

// B fragment: 32x16 bf16, lane&15 = N column; K chunks {kb..kb+8} and {kb+16..kb+24}, kb = (lane>>4)*8
__device__ __forceinline__ v16bf load_b_frag(const __bf16* lB, int lane, int nt) {
  Frag f;
  int col = nt * 16 + (lane & 15);
  int kb  = (lane >> 4) * 8;
  const uint4* p = reinterpret_cast<const uint4*>(lB + col * LDS_S + kb);
  f.q[0] = p[0];
  f.q[1] = p[2];
  return f.v;
}

// ---------------------------------------------------------------- WMMA GEMM
// MODE 0: Out = bf16(elu(A@B^T + bias))       (proj stage 1)
// MODE 1: Out = bf16(A@B^T + bias)            (proj stage 2)
// MODE 2: rs[i] += sum_j exp(2*sim)           (refl similarity)
// MODE 3: rs/cs/diag of exp(2*sim)            (betw similarity)
template <int MODE>
__global__ __launch_bounds__(256) void gemm_wmma(
    const __bf16* __restrict__ A, const __bf16* __restrict__ B, int K,
    __bf16* __restrict__ Out, const float* __restrict__ bias, int ldo,
    float* __restrict__ rs, float* __restrict__ cs, float* __restrict__ dg) {
  __shared__ __align__(16) __bf16 lA[2][128 * LDS_S];
  __shared__ __align__(16) __bf16 lB[2][128 * LDS_S];

  int m0   = blockIdx.x * 128;
  int n0   = blockIdx.y * 128;
  int lane = threadIdx.x & 31;
  int wave = threadIdx.x >> 5;
  int wrow = wave * 16;            // 8 waves x 16 rows = 128-row block
  int half = lane >> 4;

  // cooperative copy geometry: thread -> (row r, 16-elem half h) of a 128x32 panel
  int r = threadIdx.x >> 1;
  int h = threadIdx.x & 1;

  v8f zero = {0.f, 0.f, 0.f, 0.f, 0.f, 0.f, 0.f, 0.f};
  v8f acc[8];
#pragma unroll
  for (int i = 0; i < 8; i++) acc[i] = zero;

  auto issue_tile = [&](int buf, int k0) {
    const __bf16* ga = A + (size_t)(m0 + r) * K + k0 + h * 16;
    const __bf16* gb = B + (size_t)(n0 + r) * K + k0 + h * 16;
    __bf16* sa = &lA[buf][r * LDS_S + h * 16];
    __bf16* sb = &lB[buf][r * LDS_S + h * 16];
    async_b128(ga,     sa);
    async_b128(ga + 8, sa + 8);
    async_b128(gb,     sb);
    async_b128(gb + 8, sb + 8);
  };

  const int T = K >> 5;            // 32-wide K panels
  issue_tile(0, 0);                // prologue: DMA tile 0
  for (int tt = 0; tt < T; ++tt) {
    wait_async0();                 // this wave's DMA for tile tt has landed
    __syncthreads();               // everyone's DMA has landed
    if (tt + 1 < T) issue_tile((tt + 1) & 1, (tt + 1) << 5);  // overlap next DMA

    const __bf16* bufA = lA[tt & 1];
    const __bf16* bufB = lB[tt & 1];
    v16bf af   = load_a_frag(bufA, lane, wrow);
    v16bf bcur = load_b_frag(bufB, lane, 0);
#pragma unroll
    for (int nt = 0; nt < 8; nt++) {
      v16bf bnxt = bcur;
      if (nt < 7) bnxt = load_b_frag(bufB, lane, nt + 1);  // pipeline DS under WMMA
      acc[nt] = __builtin_amdgcn_wmma_f32_16x16x32_bf16(
          false, af, false, bcur, (short)0, acc[nt], false, false);
      bcur = bnxt;
    }
    __syncthreads();               // all reads of buf[tt&1] done before reuse
  }

  if (MODE <= 1) {
#pragma unroll
    for (int nt = 0; nt < 8; nt++) {
      int j = n0 + nt * 16 + (lane & 15);
      float bj = bias[j];
#pragma unroll
      for (int rr = 0; rr < 8; rr++) {
        int row = m0 + wrow + rr + 8 * half;
        float v = acc[nt][rr] + bj;
        if (MODE == 0) v = v > 0.f ? v : (__expf(v) - 1.f);   // ELU
        Out[(size_t)row * ldo + j] = (__bf16)v;
      }
    }
  } else {
    float rowacc[8] = {0.f, 0.f, 0.f, 0.f, 0.f, 0.f, 0.f, 0.f};
#pragma unroll
    for (int nt = 0; nt < 8; nt++) {
      int col = n0 + nt * 16 + (lane & 15);
      float cpart = 0.f;
#pragma unroll
      for (int rr = 0; rr < 8; rr++) {
        float e = __expf(TAU_INV * acc[nt][rr]);
        rowacc[rr] += e;
        cpart += e;
        if (MODE == 3) {
          int row = m0 + wrow + rr + 8 * half;
          if (row == col) dg[row] = e;      // diag(betw), unique writer
        }
      }
      if (MODE == 3) {
        cpart += __shfl_xor(cpart, 16);     // combine the two 16-lane halves
        if (lane < 16) atomicAdd(&cs[col], cpart);
      }
    }
#pragma unroll
    for (int rr = 0; rr < 8; rr++) {
      float v = rowacc[rr];
      v += __shfl_xor(v, 1);
      v += __shfl_xor(v, 2);
      v += __shfl_xor(v, 4);
      v += __shfl_xor(v, 8);
      if ((lane & 15) == 0) {
        int row = m0 + wrow + rr + 8 * half;
        atomicAdd(&rs[row], v);
      }
    }
  }
}

// ---------------------------------------------------------------- normalize
__global__ __launch_bounds__(256) void row_normalize(__bf16* __restrict__ H) {
  __bf16* p = H + (size_t)blockIdx.x * DIM;
  float ss = 0.f;
  for (int i = threadIdx.x; i < DIM; i += 256) {
    float v = (float)p[i];
    ss += v * v;
  }
  ss += __shfl_xor(ss, 1);  ss += __shfl_xor(ss, 2);
  ss += __shfl_xor(ss, 4);  ss += __shfl_xor(ss, 8);
  ss += __shfl_xor(ss, 16);
  __shared__ float wsum[8];
  int lane = threadIdx.x & 31, wv = threadIdx.x >> 5;
  if (lane == 0) wsum[wv] = ss;
  __syncthreads();
  if (threadIdx.x < 8) {
    float t = wsum[threadIdx.x];
    t += __shfl_xor(t, 1); t += __shfl_xor(t, 2); t += __shfl_xor(t, 4);
    if (threadIdx.x == 0) wsum[0] = rsqrtf(t);
  }
  __syncthreads();
  float rn = wsum[0];
  for (int i = threadIdx.x; i < DIM; i += 256)
    p[i] = (__bf16)((float)p[i] * rn);
}

// ---------------------------------------------------------------- loss pieces
__global__ __launch_bounds__(256) void finalize_view(
    const float* __restrict__ rs11, const float* __restrict__ rs22,
    const float* __restrict__ rs12, const float* __restrict__ cs12,
    const float* __restrict__ dg, float* __restrict__ lossAcc) {
  const float E2 = 7.38905609893065f;     // exp(sim(i,i)/tau) = e^2
  int i = blockIdx.x * 256 + threadIdx.x; // NROWS divisible by 256
  float d1 = rs11[i] + rs12[i] - E2;
  float d2 = rs22[i] + cs12[i] - E2;
  float lb = __logf(dg[i]);
  float li = 0.5f * ((__logf(d1) - lb) + (__logf(d2) - lb));
  li += __shfl_xor(li, 1);  li += __shfl_xor(li, 2);
  li += __shfl_xor(li, 4);  li += __shfl_xor(li, 8);
  li += __shfl_xor(li, 16);
  __shared__ float wsum[8];
  int lane = threadIdx.x & 31, wv = threadIdx.x >> 5;
  if (lane == 0) wsum[wv] = li;
  __syncthreads();
  if (threadIdx.x < 8) {
    float t = wsum[threadIdx.x];
    t += __shfl_xor(t, 1); t += __shfl_xor(t, 2); t += __shfl_xor(t, 4);
    if (threadIdx.x == 0) atomicAdd(lossAcc, t);
  }
}

__global__ void combine(const float* __restrict__ lossAcc,
                        const float* __restrict__ w_r1,
                        float* __restrict__ out) {
  float w = fminf(fmaxf(w_r1[0], 0.f), 1.f);
  out[0] = (w * lossAcc[0] + (1.f - w) * lossAcc[1]) / (float)NROWS;
}

// ---------------------------------------------------------------- launcher
extern "C" void kernel_launch(void* const* d_in, const int* in_sizes, int n_in,
                              void* d_out, int out_size, void* d_ws, size_t ws_size,
                              hipStream_t stream) {
  const float* ta = (const float*)d_in[0];
  const float* tb = (const float*)d_in[1];
  const float* W[4]  = {(const float*)d_in[2], (const float*)d_in[4],
                        (const float*)d_in[6], (const float*)d_in[8]};
  const float* bv[4] = {(const float*)d_in[3], (const float*)d_in[5],
                        (const float*)d_in[7], (const float*)d_in[9]};
  const float* w_r1 = (const float*)d_in[10];
  float* out = (float*)d_out;

  char* ws = (char*)d_ws;
  size_t ND = (size_t)NROWS * DIM;
  size_t DD = (size_t)DIM * DIM;
  size_t off = 0;
  __bf16* z[4];                      // z1row z2row z1col z2col
  for (int i = 0; i < 4; i++) { z[i] = (__bf16*)(ws + off); off += ND * 2; }
  __bf16* Wb[4];
  for (int i = 0; i < 4; i++) { Wb[i] = (__bf16*)(ws + off); off += DD * 2; }
  __bf16* Tbf = (__bf16*)(ws + off); off += ND * 2;
  __bf16* Hn1 = (__bf16*)(ws + off); off += ND * 2;
  __bf16* Hn2 = (__bf16*)(ws + off); off += ND * 2;
  float* sums = (float*)(ws + off);                    // 10*N + 2 floats
  float* lossAcc = sums + 10 * (size_t)NROWS;
  hipMemsetAsync(sums, 0, (10 * (size_t)NROWS + 16) * sizeof(float), stream);

  int cvtB = (int)((ND / 4 + 255) / 256);
  convert_f32_bf16<<<cvtB, 256, 0, stream>>>(ta, z[0], (int)(ND / 4));
  convert_f32_bf16<<<cvtB, 256, 0, stream>>>(tb, z[1], (int)(ND / 4));
  convert_transpose32<<<NROWS, 256, 0, stream>>>(ta, z[2]);
  convert_transpose32<<<NROWS, 256, 0, stream>>>(tb, z[3]);
  int wB = (int)((DD / 4 + 255) / 256);
  for (int i = 0; i < 4; i++)
    convert_f32_bf16<<<wB, 256, 0, stream>>>(W[i], Wb[i], (int)(DD / 4));

  dim3 blk(256);
  dim3 gProj(NROWS / 128, DIM / 128);   // 64 x 8
  dim3 gSim(NROWS / 128, NROWS / 128);  // 64 x 64

  for (int v = 0; v < 2; v++) {
    const __bf16* Z1 = z[v * 2 + 0];
    const __bf16* Z2 = z[v * 2 + 1];
    const __bf16* Wa = Wb[v * 2 + 0]; const float* ba = bv[v * 2 + 0];
    const __bf16* Wc = Wb[v * 2 + 1]; const float* bc = bv[v * 2 + 1];
    float* rs11 = sums + (size_t)v * 5 * NROWS;
    float* rs22 = rs11 + NROWS;
    float* rs12 = rs11 + 2 * NROWS;
    float* cs12 = rs11 + 3 * NROWS;
    float* dg   = rs11 + 4 * NROWS;

    gemm_wmma<0><<<gProj, blk, 0, stream>>>(Z1, Wa, DIM, Tbf, ba, DIM,
                                            nullptr, nullptr, nullptr);
    gemm_wmma<1><<<gProj, blk, 0, stream>>>(Tbf, Wc, DIM, Hn1, bc, DIM,
                                            nullptr, nullptr, nullptr);
    gemm_wmma<0><<<gProj, blk, 0, stream>>>(Z2, Wa, DIM, Tbf, ba, DIM,
                                            nullptr, nullptr, nullptr);
    gemm_wmma<1><<<gProj, blk, 0, stream>>>(Tbf, Wc, DIM, Hn2, bc, DIM,
                                            nullptr, nullptr, nullptr);
    row_normalize<<<NROWS, 256, 0, stream>>>(Hn1);
    row_normalize<<<NROWS, 256, 0, stream>>>(Hn2);
    gemm_wmma<2><<<gSim, blk, 0, stream>>>(Hn1, Hn1, DIM, nullptr, nullptr, 0,
                                           rs11, nullptr, nullptr);
    gemm_wmma<2><<<gSim, blk, 0, stream>>>(Hn2, Hn2, DIM, nullptr, nullptr, 0,
                                           rs22, nullptr, nullptr);
    gemm_wmma<3><<<gSim, blk, 0, stream>>>(Hn1, Hn2, DIM, nullptr, nullptr, 0,
                                           rs12, cs12, dg);
    finalize_view<<<NROWS / 256, 256, 0, stream>>>(rs11, rs22, rs12, cs12, dg,
                                                   lossAcc + v);
  }
  combine<<<1, 1, 0, stream>>>(lossAcc, w_r1, out);
}
